// TransformerBlock_52372831207675
// MI455X (gfx1250) — matrix operand order repaired and checked by reading, not compiled, not run on hardware
//
#include <hip/hip_runtime.h>
#include <math.h>

typedef __attribute__((ext_vector_type(16))) _Float16 v16h;
typedef __attribute__((ext_vector_type(8)))  _Float16 v8h;
typedef __attribute__((ext_vector_type(4)))  _Float16 v4h;
typedef __attribute__((ext_vector_type(8)))  float    v8f;

#define CH 64

// hardware transcendental path: v_exp_f32
__device__ __forceinline__ float siluf(float x) { return x / (1.f + __expf(-x)); }

// ---- WMMA fragment loaders (CDNA5 16x16x32 f16 layouts, ISA 7.12.2) ----
// A: 16x32 f16. Lane L holds row M=L%16; halves are two contiguous 8-half runs:
//    k = kb + (L>=16?8:0) + {0..7}  and  + 16 + {0..7}
__device__ __forceinline__ v16h ldsA(const _Float16* t, int ld, int lane, int kb) {
  int m  = lane & 15;
  int kh = (lane < 16) ? 0 : 8;
  const _Float16* base = t + m * ld + kb + kh;
  v8h lo = *(const v8h*)(base);       // 16B aligned
  v8h hi = *(const v8h*)(base + 16);
  return __builtin_shufflevector(lo, hi, 0,1,2,3,4,5,6,7,8,9,10,11,12,13,14,15);
}
// B: 32x16 f16 read from TRANSPOSED weights wT[n][k] (two 8-half runs):
//    lane L holds col N=nb+L%16; halves i<8 hold K = kb + (L>=16?8:0) + i, i>=8 hold + 16 + (i-8)
__device__ __forceinline__ v16h ldsBT(const _Float16* wT, int lane, int kb, int nb) {
  int n  = nb + (lane & 15);
  int ko = kb + ((lane < 16) ? 0 : 8);
  const _Float16* base = wT + n * CH + ko;  // 16B aligned
  v8h lo = *(const v8h*)(base);
  v8h hi = *(const v8h*)(base + 16);
  return __builtin_shufflevector(lo, hi, 0,1,2,3,4,5,6,7,8,9,10,11,12,13,14,15);
}
// D/C (v8f): element j is row M = j + (lane>=16?8:0), col N = ntile*16 + lane%16

// out(16x64) += A(16x64, f16 LDS row-major ld) * W(64x64, f16 LDS stored transposed)
__device__ __forceinline__ void gemm_16x64_64(const _Float16* atile, int lda,
                                              const _Float16* wT, v8f acc[4], int lane) {
#pragma unroll
  for (int kt = 0; kt < 2; ++kt) {
    v16h a = ldsA(atile, lda, lane, kt * 32);
#pragma unroll
    for (int nt = 0; nt < 4; ++nt) {
      v16h b = ldsBT(wT, lane, kt * 32, nt * 16);
      acc[nt] = __builtin_amdgcn_wmma_f32_16x16x32_f16(
          false, a, false, b, (short)0, acc[nt], false, false);
    }
  }
}

__device__ __forceinline__ void atomicMaxF(float* addr, float v) {
  if (v >= 0.f) atomicMax((int*)addr, __float_as_int(v));
  else          atomicMin((unsigned int*)addr, __float_as_uint(v));
}

// ---------------- init: -inf max, zero denom/accum/stats ----------------
__global__ void k_init(unsigned int* mx, float* den, float* accb, float* stats, long long n) {
  long long i = (long long)blockIdx.x * blockDim.x + threadIdx.x;
  if (i < n) { mx[i] = 0xFF800000u; den[i] = 0.f; accb[i] = 0.f; }
  if (i < 512) stats[i] = 0.f;
}

// ---------------- Y = X @ W + bias (M x 64 x 64) ----------------
__global__ __launch_bounds__(256) void k_gemm_bias(const float* __restrict__ X,
                                                   const float* __restrict__ W,
                                                   const float* __restrict__ bias,
                                                   float* __restrict__ Y, int nrows) {
  __shared__ __align__(16) _Float16 sX[128 * CH];
  __shared__ __align__(16) _Float16 sW[CH * CH];   // transposed: sW[n*64+k]
  int tid = threadIdx.x;
  for (int i = tid; i < CH * CH; i += 256) sW[(i & 63) * CH + (i >> 6)] = (_Float16)W[i];
  int rb = blockIdx.x * 128;
  for (int i = tid; i < 128 * CH; i += 256) {
    int r = rb + (i >> 6);
    sX[i] = (_Float16)((r < nrows) ? X[(long long)r * CH + (i & 63)] : 0.f);
  }
  __syncthreads();
  int wv = tid >> 5, lane = tid & 31;
  v8f acc[4] = {};
  gemm_16x64_64(&sX[wv * 16 * CH], CH, sW, acc, lane);
  int mbase = rb + wv * 16 + ((lane < 16) ? 0 : 8);
  int nlo = lane & 15;
  float b4[4];
#pragma unroll
  for (int nt = 0; nt < 4; ++nt) b4[nt] = bias[nt * 16 + nlo];
  if (rb + 128 <= nrows) {   // full tile, no guards
#pragma unroll
    for (int nt = 0; nt < 4; ++nt) {
      int col = nt * 16 + nlo;
#pragma unroll
      for (int j = 0; j < 8; ++j)
        Y[(long long)(mbase + j) * CH + col] = acc[nt][j] + b4[nt];
    }
  } else {
#pragma unroll
    for (int nt = 0; nt < 4; ++nt) {
      int col = nt * 16 + nlo;
#pragma unroll
      for (int j = 0; j < 8; ++j) {
        int r = mbase + j;
        if (r < nrows) Y[(long long)r * CH + col] = acc[nt][j] + b4[nt];
      }
    }
  }
}

// ---------------- per-channel sum / sumsq ----------------
__global__ __launch_bounds__(256) void k_stats(const float* __restrict__ Y,
                                               float* __restrict__ S, int nrows) {
  __shared__ float sh[2][4][64];
  int tid = threadIdx.x;
  int c = tid & 63, g = tid >> 6;
  float s = 0.f, ss = 0.f;
  int r0 = blockIdx.x * 256;
  for (int r = r0 + g; r < r0 + 256 && r < nrows; r += 4) {
    float v = Y[(long long)r * CH + c];
    s += v; ss += v * v;
  }
  sh[0][g][c] = s; sh[1][g][c] = ss;
  __syncthreads();
  if (g == 0) {
    atomicAdd(&S[c],      sh[0][0][c] + sh[0][1][c] + sh[0][2][c] + sh[0][3][c]);
    atomicAdd(&S[64 + c], sh[1][0][c] + sh[1][1][c] + sh[1][2][c] + sh[1][3][c]);
  }
}

// ---------------- fold BN into scale/shift: y = h*a + b ----------------
__global__ void k_finalize(const float* __restrict__ S, const float* __restrict__ gamma,
                           const float* __restrict__ beta, float* __restrict__ AB, float nInv) {
  int c = threadIdx.x;
  if (c >= 64) return;
  float mu  = S[c] * nInv;
  float var = S[64 + c] * nInv - mu * mu;
  float a   = gamma[c] * rsqrtf(var + 1e-5f);
  AB[c] = a;
  AB[64 + c] = beta[c] - mu * a;
}

// ---------------- xh = silu(h*a+b); V=xh@Wv K=xh@Wk Q=xh@Wq ----------------
__global__ __launch_bounds__(256) void k_vkq(const float* __restrict__ H,
                                             const float* __restrict__ AB,
                                             const float* __restrict__ Wv,
                                             const float* __restrict__ Wk,
                                             const float* __restrict__ Wq,
                                             float* __restrict__ V, float* __restrict__ K,
                                             float* __restrict__ Q, int nrows) {
  __shared__ __align__(16) _Float16 sX[128 * CH];
  __shared__ __align__(16) _Float16 sW[3][CH * CH];  // transposed
  int tid = threadIdx.x;
  for (int i = tid; i < CH * CH; i += 256) {
    int tj = (i & 63) * CH + (i >> 6);
    sW[0][tj] = (_Float16)Wv[i];
    sW[1][tj] = (_Float16)Wk[i];
    sW[2][tj] = (_Float16)Wq[i];
  }
  int rb = blockIdx.x * 128;
  for (int i = tid; i < 128 * CH; i += 256) {
    int r = rb + (i >> 6), c = i & 63;
    float h = (r < nrows) ? H[(long long)r * CH + c] : 0.f;
    sX[i] = (_Float16)siluf(h * AB[c] + AB[64 + c]);
  }
  __syncthreads();
  int wv = tid >> 5, lane = tid & 31;
  int mbase = rb + wv * 16 + ((lane < 16) ? 0 : 8);
  int nlo = lane & 15;
  bool full = (rb + 128 <= nrows);
  float* outs[3] = {V, K, Q};
#pragma unroll
  for (int m = 0; m < 3; ++m) {
    v8f acc[4] = {};
    gemm_16x64_64(&sX[wv * 16 * CH], CH, sW[m], acc, lane);
    if (full) {
#pragma unroll
      for (int nt = 0; nt < 4; ++nt) {
        int col = nt * 16 + nlo;
#pragma unroll
        for (int j = 0; j < 8; ++j)
          outs[m][(long long)(mbase + j) * CH + col] = acc[nt][j];
      }
    } else {
#pragma unroll
      for (int nt = 0; nt < 4; ++nt) {
        int col = nt * 16 + nlo;
#pragma unroll
        for (int j = 0; j < 8; ++j) {
          int r = mbase + j;
          if (r < nrows) outs[m][(long long)r * CH + col] = acc[nt][j];
        }
      }
    }
  }
}

// ---------------- edge kernel: delta, logits (+atomic max) ----------------
__global__ __launch_bounds__(256) void k_edges(
    const float* __restrict__ pos, const int* __restrict__ esrc, const int* __restrict__ edst,
    const float* __restrict__ Qb, const float* __restrict__ Kb,
    const float* __restrict__ P1, const float* __restrict__ pb1,
    const float* __restrict__ P2, const float* __restrict__ pb2,
    const float* __restrict__ A1, const float* __restrict__ ab1,
    const float* __restrict__ A2, const float* __restrict__ ab2,
    _Float16* __restrict__ Dout, _Float16* __restrict__ Lout, float* __restrict__ MX,
    long long E, long long Etot) {
  __shared__ __align__(16) _Float16 sP2[CH * CH], sA1[CH * CH], sA2[CH * CH]; // transposed
  __shared__ float sP1[3 * CH];
  __shared__ float sb[4 * CH];                       // pb1 | pb2 | ab1 | ab2
  __shared__ __align__(16) _Float16 scr[8][16 * CH]; // per-wave 16x64 f16 scratch
  __shared__ int sS[8][16], sD[8][16];
  __shared__ float sPD[8][16][3];

  int tid = threadIdx.x, wv = tid >> 5, lane = tid & 31;
  for (int i = tid; i < CH * CH; i += 256) {
    int tj = (i & 63) * CH + (i >> 6);
    sP2[tj] = (_Float16)P2[i];
    sA1[tj] = (_Float16)A1[i];
    sA2[tj] = (_Float16)A2[i];
  }
  if (tid < 3 * CH) sP1[tid] = P1[tid];
  if (tid < CH) {
    sb[tid] = pb1[tid]; sb[64 + tid] = pb2[tid];
    sb[128 + tid] = ab1[tid]; sb[192 + tid] = ab2[tid];
  }
  long long eb = (long long)blockIdx.x * 128 + (long long)wv * 16;
  bool full = (eb + 16 <= Etot);
  if (lane < 16) {
    long long e = eb + lane;
    int s = 0, d = 0;
    if (e < Etot) {
      if (e < E) { s = esrc[e]; d = edst[e]; }
      else       { s = d = (int)(e - E); }
    }
    sS[wv][lane] = s; sD[wv][lane] = d;
    sPD[wv][lane][0] = pos[(long long)d * 3 + 0] - pos[(long long)s * 3 + 0];
    sPD[wv][lane][1] = pos[(long long)d * 3 + 1] - pos[(long long)s * 3 + 1];
    sPD[wv][lane][2] = pos[(long long)d * 3 + 2] - pos[(long long)s * 3 + 2];
  }
  __syncthreads();

  // lane-pair mapping: lane -> row me2 = lane/2, channels c0..c0+31 (contiguous)
  int me2 = lane >> 1, c0 = (lane & 1) * 32;

  // stage 1: delta1 = silu(pd @ P1 + pb1)   (K=3, plain VALU, vectorizable)
  {
    float p0 = sPD[wv][me2][0], p1 = sPD[wv][me2][1], p2 = sPD[wv][me2][2];
#pragma unroll
    for (int j = 0; j < 32; ++j) {
      int c = c0 + j;
      float v = p0 * sP1[c] + p1 * sP1[64 + c] + p2 * sP1[128 + c] + sb[c];
      scr[wv][me2 * CH + c] = (_Float16)siluf(v);
    }
  }
  __syncthreads();

  int rowoff = (lane < 16) ? 0 : 8, nlo = lane & 15;

  // stage 2: delta = silu(delta1 @ P2 + pb2)
  {
    v8f acc[4] = {};
    gemm_16x64_64(scr[wv], CH, sP2, acc, lane);
    __syncthreads();
    if (full) {
#pragma unroll
      for (int nt = 0; nt < 4; ++nt) {
        int c = nt * 16 + nlo;
#pragma unroll
        for (int j = 0; j < 8; ++j) {
          int me = j + rowoff;
          float de = siluf(acc[nt][j] + sb[64 + c]);
          scr[wv][me * CH + c] = (_Float16)de;
          Dout[(eb + me) * CH + c] = (_Float16)de;
        }
      }
    } else {
#pragma unroll
      for (int nt = 0; nt < 4; ++nt) {
        int c = nt * 16 + nlo;
#pragma unroll
        for (int j = 0; j < 8; ++j) {
          int me = j + rowoff;
          float de = siluf(acc[nt][j] + sb[64 + c]);
          scr[wv][me * CH + c] = (_Float16)de;
          if (eb + me < Etot) Dout[(eb + me) * CH + c] = (_Float16)de;
        }
      }
    }
  }
  __syncthreads();

  // stage 3: alpha_pre = q[dst] - k[src] + delta  (lane-pair, float4 gathers,
  // in-place b64 RMW on disjoint addresses)
  {
    int s = sS[wv][me2], d = sD[wv][me2];
    const float4* qp = (const float4*)&Qb[(long long)d * CH + c0];
    const float4* kp = (const float4*)&Kb[(long long)s * CH + c0];
    v4h* sp = (v4h*)&scr[wv][me2 * CH + c0];
#pragma unroll
    for (int j = 0; j < 8; ++j) {
      float4 q4 = qp[j];
      float4 k4 = kp[j];
      v4h de = sp[j];
      v4h o;
      o[0] = (_Float16)(q4.x - k4.x + (float)de[0]);
      o[1] = (_Float16)(q4.y - k4.y + (float)de[1]);
      o[2] = (_Float16)(q4.z - k4.z + (float)de[2]);
      o[3] = (_Float16)(q4.w - k4.w + (float)de[3]);
      sp[j] = o;
    }
  }
  __syncthreads();

  // stage 4: a1 = silu(alpha_pre @ A1 + ab1)
  {
    v8f acc[4] = {};
    gemm_16x64_64(scr[wv], CH, sA1, acc, lane);
    __syncthreads();
#pragma unroll
    for (int nt = 0; nt < 4; ++nt) {
      int c = nt * 16 + nlo;
#pragma unroll
      for (int j = 0; j < 8; ++j) {
        int me = j + rowoff;
        scr[wv][me * CH + c] = (_Float16)siluf(acc[nt][j] + sb[128 + c]);
      }
    }
  }
  __syncthreads();

  // stage 5: logits = silu(a1 @ A2 + ab2); store + segment atomic max
  {
    v8f acc[4] = {};
    gemm_16x64_64(scr[wv], CH, sA2, acc, lane);
    if (full) {
#pragma unroll
      for (int nt = 0; nt < 4; ++nt) {
        int c = nt * 16 + nlo;
#pragma unroll
        for (int j = 0; j < 8; ++j) {
          int me = j + rowoff;
          float lg = siluf(acc[nt][j] + sb[192 + c]);
          Lout[(eb + me) * CH + c] = (_Float16)lg;
          atomicMaxF(&MX[(long long)sD[wv][me] * CH + c], lg);
        }
      }
    } else {
#pragma unroll
      for (int nt = 0; nt < 4; ++nt) {
        int c = nt * 16 + nlo;
#pragma unroll
        for (int j = 0; j < 8; ++j) {
          int me = j + rowoff;
          if (eb + me < Etot) {
            float lg = siluf(acc[nt][j] + sb[192 + c]);
            Lout[(eb + me) * CH + c] = (_Float16)lg;
            atomicMaxF(&MX[(long long)sD[wv][me] * CH + c], lg);
          }
        }
      }
    }
  }
}

// ---------------- pass A: denom = segment_sum(exp(logit - m)) ----------------
__global__ __launch_bounds__(256) void k_denom(const _Float16* __restrict__ L,
                                               const float* __restrict__ MX,
                                               float* __restrict__ DEN,
                                               const int* __restrict__ edst,
                                               long long E, long long Etot) {
  long long t = (long long)blockIdx.x * 256 + threadIdx.x;
  long long e = t >> 5;
  if (e >= Etot) return;
  int c0 = (int)(t & 31) * 2;
  int d = (e < E) ? edst[e] : (int)(e - E);
#pragma unroll
  for (int u = 0; u < 2; ++u) {
    int c = c0 + u;
    float ex = __expf((float)L[e * CH + c] - MX[(long long)d * CH + c]);
    atomicAdd(&DEN[(long long)d * CH + c], ex);
  }
}

// ---------------- pass B: acc += attn * (v[src] + delta) ----------------
__global__ __launch_bounds__(256) void k_scatter(const _Float16* __restrict__ L,
                                                 const _Float16* __restrict__ DE,
                                                 const float* __restrict__ MX,
                                                 const float* __restrict__ DEN,
                                                 const float* __restrict__ Vb,
                                                 float* __restrict__ ACC,
                                                 const int* __restrict__ esrc,
                                                 const int* __restrict__ edst,
                                                 long long E, long long Etot) {
  long long t = (long long)blockIdx.x * 256 + threadIdx.x;
  long long e = t >> 5;
  if (e >= Etot) return;
  int c0 = (int)(t & 31) * 2;
  int s, d;
  if (e < E) { s = esrc[e]; d = edst[e]; }
  else       { s = d = (int)(e - E); }
#pragma unroll
  for (int u = 0; u < 2; ++u) {
    int c = c0 + u;
    float w = __expf((float)L[e * CH + c] - MX[(long long)d * CH + c])
            / DEN[(long long)d * CH + c];
    float msg = Vb[(long long)s * CH + c] + (float)DE[e * CH + c];
    atomicAdd(&ACC[(long long)d * CH + c], w * msg);
  }
}

// ---------------- final BN apply ----------------
__global__ void k_apply(const float* __restrict__ Y, const float* __restrict__ AB,
                        float* __restrict__ O, long long n) {
  long long i = (long long)blockIdx.x * blockDim.x + threadIdx.x;
  if (i < n) {
    int c = (int)(i & 63);
    O[i] = Y[i] * AB[c] + AB[64 + c];
  }
}

extern "C" void kernel_launch(void* const* d_in, const int* in_sizes, int n_in,
                              void* d_out, int out_size, void* d_ws, size_t ws_size,
                              hipStream_t stream) {
  (void)n_in; (void)out_size; (void)ws_size;
  const float* x    = (const float*)d_in[0];
  const float* pos  = (const float*)d_in[1];
  const int*   ei   = (const int*)d_in[2];
  const float* Win  = (const float*)d_in[3];
  const float* bin_ = (const float*)d_in[4];
  const float* gin  = (const float*)d_in[5];
  const float* bein = (const float*)d_in[6];
  const float* Wv   = (const float*)d_in[7];
  const float* Wk   = (const float*)d_in[8];
  const float* Wq   = (const float*)d_in[9];
  const float* P1   = (const float*)d_in[10];
  const float* pb1  = (const float*)d_in[11];
  const float* P2   = (const float*)d_in[12];
  const float* pb2  = (const float*)d_in[13];
  const float* A1   = (const float*)d_in[14];
  const float* ab1  = (const float*)d_in[15];
  const float* A2   = (const float*)d_in[16];
  const float* ab2  = (const float*)d_in[17];
  const float* Wout = (const float*)d_in[18];
  const float* bout = (const float*)d_in[19];
  const float* gout = (const float*)d_in[20];
  const float* beout= (const float*)d_in[21];

  const long long N    = in_sizes[0] / CH;
  const long long E    = in_sizes[2] / 2;
  const long long Etot = E + N;
  const int* esrc = ei;
  const int* edst = ei + E;

  // workspace carve-up
  char* ws = (char*)d_ws;
  size_t off = 0;
  auto carve = [&](size_t bytes) { char* p = ws + off; off += (bytes + 255) & ~(size_t)255; return p; };
  float*    h    = (float*)carve(N * CH * 4);   // reused as y for the output GEMM
  float*    Vb   = (float*)carve(N * CH * 4);
  float*    Kb   = (float*)carve(N * CH * 4);
  float*    Qb   = (float*)carve(N * CH * 4);
  float*    MX   = (float*)carve(N * CH * 4);
  float*    DEN  = (float*)carve(N * CH * 4);
  float*    ACC  = (float*)carve(N * CH * 4);
  float*    stats= (float*)carve(512 * 4);      // [0:128)=stats1, [128:256)=stats2
  float*    AB1  = (float*)carve(128 * 4);
  float*    AB2  = (float*)carve(128 * 4);
  _Float16* DE   = (_Float16*)carve(Etot * CH * 2);
  _Float16* LO   = (_Float16*)carve(Etot * CH * 2);

  const int TB = 256;
  long long nNC = N * CH;
  int gInit  = (int)((nNC + TB - 1) / TB);
  int gGemm  = (int)((N + 127) / 128);
  int gStats = (int)((N + 255) / 256);
  int gEdge  = (int)((Etot + 127) / 128);
  int gEdgeC = (int)((Etot * 32 + TB - 1) / TB);
  float nInv = 1.f / (float)N;

  k_init<<<gInit, TB, 0, stream>>>((unsigned int*)MX, DEN, ACC, stats, nNC);
  // mlp_in linear
  k_gemm_bias<<<gGemm, TB, 0, stream>>>(x, Win, bin_, h, (int)N);
  // BN1 stats + fold
  k_stats<<<gStats, TB, 0, stream>>>(h, stats, (int)N);
  k_finalize<<<1, 64, 0, stream>>>(stats, gin, bein, AB1, nInv);
  // silu(bn(h)) -> v,k,q
  k_vkq<<<gGemm, TB, 0, stream>>>(h, AB1, Wv, Wk, Wq, Vb, Kb, Qb, (int)N);
  // per-edge MLPs (WMMA) + segment max
  k_edges<<<gEdge, TB, 0, stream>>>(pos, esrc, edst, Qb, Kb, P1, pb1, P2, pb2,
                                    A1, ab1, A2, ab2, DE, LO, MX, E, Etot);
  // softmax denom, then weighted scatter
  k_denom<<<gEdgeC, TB, 0, stream>>>(LO, MX, DEN, edst, E, Etot);
  k_scatter<<<gEdgeC, TB, 0, stream>>>(LO, DE, MX, DEN, Vb, ACC, esrc, edst, E, Etot);
  // mlp_out linear (reuse h as y) + BN2
  k_gemm_bias<<<gGemm, TB, 0, stream>>>(ACC, Wout, bout, h, (int)N);
  k_stats<<<gStats, TB, 0, stream>>>(h, stats + 128, (int)N);
  k_finalize<<<1, 64, 0, stream>>>(stats + 128, gout, beout, AB2, nInv);
  k_apply<<<gInit, TB, 0, stream>>>(h, AB2, (float*)d_out, nNC);
}
